// ModulatedConvBlock_43576738185306
// MI455X (gfx1250) — matrix operand order repaired and checked by reading, not compiled, not run on hardware
//
#include <hip/hip_runtime.h>
#include <hip/hip_bf16.h>

typedef __attribute__((ext_vector_type(16))) _Float16 v16h;
typedef __attribute__((ext_vector_type(8)))  _Float16 v8h;
typedef __attribute__((ext_vector_type(8)))  float    v8f;

union Frag16 { v16h v; v8h h[2]; };

#define CIN   512
#define COUT  512
#define BATCH 8
#define HOUT  64        // 2*H
#define PADW  66        // HOUT + 2 halo
#define EPSV  1e-8f
#define STYLE_SCALE 0.0625f   // sqrt(2/512)
#define OUT_SCALE   0.0625f   // sqrt(2/512)
#define INV_ROOT2   (1.0f / 1.41421f)

// ---------------------------------------------------------------------------
// 1) style_std[b,i] = (style_base[b,:] * STYLE_SCALE) . affine_w[i,:] + affine_b[i] + 1
__global__ void style_kernel(const float* __restrict__ sb,
                             const float* __restrict__ aw,
                             const float* __restrict__ ab,
                             float* __restrict__ style_std) {
  int idx = blockIdx.x * blockDim.x + threadIdx.x;   // 8*512
  int b = idx >> 9, i = idx & 511;
  const float* s = sb + b * CIN;
  const float* w = aw + (size_t)i * CIN;
  float acc = 0.f;
  for (int k = 0; k < CIN; ++k) acc += s[k] * w[k];
  style_std[idx] = acc * STYLE_SCALE + ab[i] + 1.0f;
}

// ---------------------------------------------------------------------------
// 2a) wsq[o,i] = sum_k w[o,i,k]^2  ;  repack weights f32 -> f16 as wk[tap][o][i]
__global__ void prep_w_kernel(const float* __restrict__ cw,
                              float* __restrict__ wsq,
                              _Float16* __restrict__ wk) {
  int idx = blockIdx.x * blockDim.x + threadIdx.x;   // 512*512 (o*512+i)
  const float* w = cw + (size_t)idx * 9;
  float s = 0.f;
#pragma unroll
  for (int k = 0; k < 9; ++k) {
    float v = w[k];
    s += v * v;
    wk[(size_t)k * (COUT * CIN) + idx] = (_Float16)v;
  }
  wsq[idx] = s;
}

// 2b) demod[b,o] = rsqrt( sum_i wsq[o,i] * style_std[b,i]^2 + eps )
__global__ void demod_kernel(const float* __restrict__ wsq,
                             const float* __restrict__ style_std,
                             float* __restrict__ demod) {
  int idx = blockIdx.x * blockDim.x + threadIdx.x;   // 8*512 (b*512+o)
  int b = idx >> 9, o = idx & 511;
  const float* wr = wsq + (size_t)o * CIN;
  const float* sr = style_std + b * CIN;
  float acc = EPSV;
  for (int i = 0; i < CIN; ++i) { float s = sr[i]; acc += wr[i] * s * s; }
  demod[idx] = rsqrtf(acc);
}

// ---------------------------------------------------------------------------
// 3) bilinear x2 upsample (half-pixel) * style, stored channel-last f16 with
//    a 1-pixel zero halo: xm[b][py 0..65][px 0..65][i 0..511]
__global__ void upsample_mod_kernel(const float* __restrict__ x,
                                    const float* __restrict__ style,
                                    _Float16* __restrict__ xm) {
  int blk = blockIdx.x;                 // 8*66*66 blocks, 128 threads
  int px = blk % PADW; blk /= PADW;
  int py = blk % PADW; int b = blk / PADW;
  _Float16* dst = xm + ((((size_t)b * PADW + py) * PADW + px) << 9);
  if (px == 0 || px == PADW - 1 || py == 0 || py == PADW - 1) {
    for (int i = threadIdx.x; i < CIN; i += 128) dst[i] = (_Float16)0.0f;
    return;
  }
  int oy = py - 1, ox = px - 1;
  int ky = oy >> 1, kx = ox >> 1;
  int y0, y1, x0, x1; float wy0, wy1, wx0, wx1;
  if (oy & 1) { y0 = ky;                    y1 = (ky + 1 > 31) ? 31 : ky + 1; wy0 = 0.75f; wy1 = 0.25f; }
  else        { y0 = (ky - 1 < 0) ? 0 : ky - 1; y1 = ky;                     wy0 = 0.25f; wy1 = 0.75f; }
  if (ox & 1) { x0 = kx;                    x1 = (kx + 1 > 31) ? 31 : kx + 1; wx0 = 0.75f; wx1 = 0.25f; }
  else        { x0 = (kx - 1 < 0) ? 0 : kx - 1; x1 = kx;                     wx0 = 0.25f; wx1 = 0.75f; }
  const float* xb = x + (size_t)b * CIN * 1024;
  const float* sr = style + b * CIN;
  for (int i = threadIdx.x; i < CIN; i += 128) {
    const float* xc = xb + (size_t)i * 1024;
    float v = wy0 * (wx0 * xc[y0 * 32 + x0] + wx1 * xc[y0 * 32 + x1])
            + wy1 * (wx0 * xc[y1 * 32 + x0] + wx1 * xc[y1 * 32 + x1]);
    dst[i] = (_Float16)(v * sr[i]);
  }
}

// ---------------------------------------------------------------------------
// 4) main conv: 9 shifted GEMMs via v_wmma_f32_16x16x32_f16, fused epilogue.
//    One wave computes 64 output channels x 64 x-pixels of one (b, yy) row.
//    (4x4 register tile: A reuse x4, B reuse x4 -> 16 loads per 16 WMMAs.)
//    All 8 waves of a block share the same row -> B fragments hit in WGP$.
__global__ __launch_bounds__(256)
void conv_wmma_kernel(const _Float16* __restrict__ wk,
                      const _Float16* __restrict__ xm,
                      const float* __restrict__ demod,
                      const float* __restrict__ conv_b,
                      const float* __restrict__ noise_scalar,
                      const float* __restrict__ noise,
                      const float* __restrict__ t,
                      const float* __restrict__ prelu_a,
                      float* __restrict__ yout) {
  const int lane  = threadIdx.x & 31;
  const int wave  = threadIdx.x >> 5;
  const int gw    = blockIdx.x * 8 + wave;   // 4096 waves
  const int og    = gw & 7;                  // o-group of 64 channels
  const int row   = gw >> 3;                 // 0..511 (same for all 8 waves/block)
  const int b     = row >> 6;
  const int yy    = row & 63;
  const int laneL = lane & 15;
  const int laneH = lane >> 4;

  v8f c[4][4];                               // [o-tile][n-tile]
#pragma unroll
  for (int i = 0; i < 4; ++i)
#pragma unroll
    for (int j = 0; j < 4; ++j)
#pragma unroll
      for (int e = 0; e < 8; ++e) c[i][j][e] = 0.0f;

  size_t aRow[4];
  aRow[0] = (size_t)((og * 64 + laneL) * CIN + laneH * 8);
#pragma unroll
  for (int ot = 1; ot < 4; ++ot) aRow[ot] = aRow[ot - 1] + (size_t)16 * CIN;

#pragma unroll
  for (int tap = 0; tap < 9; ++tap) {
    const int dy = tap / 3 - 1;
    const int dx = tap % 3 - 1;
    const _Float16* wkt = wk + (size_t)tap * (COUT * CIN);
    const size_t rb = ((size_t)(b * PADW + yy + dy + 1)) * PADW;
    size_t bBase[4];
#pragma unroll
    for (int nt = 0; nt < 4; ++nt)
      bBase[nt] = (rb + (size_t)(nt * 16 + laneL + dx + 1)) * CIN + laneH * 8;

    for (int k0 = 0; k0 < CIN; k0 += 32) {
      Frag16 a[4];
#pragma unroll
      for (int ot = 0; ot < 4; ++ot) {
        a[ot].h[0] = *(const v8h*)(wkt + aRow[ot] + k0);
        a[ot].h[1] = *(const v8h*)(wkt + aRow[ot] + k0 + 16);
      }
#pragma unroll
      for (int nt = 0; nt < 4; ++nt) {
        Frag16 bf;
        bf.h[0] = *(const v8h*)(xm + bBase[nt] + k0);
        bf.h[1] = *(const v8h*)(xm + bBase[nt] + k0 + 16);
#pragma unroll
        for (int ot = 0; ot < 4; ++ot) {
          c[ot][nt] = __builtin_amdgcn_wmma_f32_16x16x32_f16(
              false, a[ot].v, false, bf.v, (short)0, c[ot][nt], false, false);
        }
      }
    }
  }

  // Epilogue: demod, bias, noise, PReLU, skip add, 1/sqrt(2)
  const float pa = prelu_a[0];
  float nz[4];
#pragma unroll
  for (int nt = 0; nt < 4; ++nt)
    nz[nt] = noise[(size_t)b * 4096 + yy * 64 + nt * 16 + laneL];

#pragma unroll
  for (int ot = 0; ot < 4; ++ot) {
#pragma unroll
    for (int r = 0; r < 8; ++r) {
      const int o  = og * 64 + ot * 16 + laneH * 8 + r;
      const float dm = demod[b * COUT + o];
      const float cb = conv_b[o];
      const float ns = noise_scalar[o];
      const size_t yrow = (((size_t)(b * COUT + o) * 64 + yy) * 64);
#pragma unroll
      for (int nt = 0; nt < 4; ++nt) {
        const int x = nt * 16 + laneL;
        float v = c[ot][nt][r] * dm + cb + ns * nz[nt];
        v = (v >= 0.f) ? v : pa * v;
        v = (v + t[yrow + x]) * INV_ROOT2;
        yout[yrow + x] = v;
      }
    }
  }
}

// ---------------------------------------------------------------------------
// 5) toRGB 1x1 conv + clamp: out[b,c,p] = clip(sum_o y[b,o,p]*OUT_SCALE*w[c,o] + b[c])
__global__ void torgb_kernel(const float* __restrict__ y,
                             const float* __restrict__ out_w,
                             const float* __restrict__ out_b,
                             float* __restrict__ out) {
  int p = blockIdx.x * blockDim.x + threadIdx.x;  // 8*4096
  int b = p >> 12, pix = p & 4095;
  const float* yb = y + (size_t)b * COUT * 4096 + pix;
  float a0 = 0.f, a1 = 0.f, a2 = 0.f;
  for (int o = 0; o < COUT; ++o) {
    float v = yb[(size_t)o * 4096];
    a0 += v * out_w[o];
    a1 += v * out_w[COUT + o];
    a2 += v * out_w[2 * COUT + o];
  }
  a0 = fminf(fmaxf(a0 * OUT_SCALE + out_b[0], 0.f), 1.f);
  a1 = fminf(fmaxf(a1 * OUT_SCALE + out_b[1], 0.f), 1.f);
  a2 = fminf(fmaxf(a2 * OUT_SCALE + out_b[2], 0.f), 1.f);
  out[((size_t)b * 3 + 0) * 4096 + pix] = a0;
  out[((size_t)b * 3 + 1) * 4096 + pix] = a1;
  out[((size_t)b * 3 + 2) * 4096 + pix] = a2;
}

// ---------------------------------------------------------------------------
extern "C" void kernel_launch(void* const* d_in, const int* in_sizes, int n_in,
                              void* d_out, int out_size, void* d_ws, size_t ws_size,
                              hipStream_t stream) {
  (void)in_sizes; (void)n_in; (void)out_size; (void)ws_size;

  const float* x     = (const float*)d_in[0];
  const float* sb    = (const float*)d_in[1];
  const float* noise = (const float*)d_in[2];
  const float* t     = (const float*)d_in[3];
  const float* aw    = (const float*)d_in[4];
  const float* ab    = (const float*)d_in[5];
  const float* cw    = (const float*)d_in[6];
  const float* cb    = (const float*)d_in[7];
  const float* pa    = (const float*)d_in[8];
  const float* ns    = (const float*)d_in[9];
  const float* ow    = (const float*)d_in[10];
  const float* ob    = (const float*)d_in[11];

  float* yout = (float*)d_out;
  float* out  = yout + (size_t)BATCH * COUT * HOUT * HOUT;

  // Workspace carve-up (all offsets 256B-aligned), total ~41.4 MB
  char* ws = (char*)d_ws;
  float*    style_std = (float*)ws;                                   // 16 KB
  float*    demod     = (float*)(ws + 16384);                         // 16 KB
  float*    wsq       = (float*)(ws + 32768);                         // 1 MB
  _Float16* wk        = (_Float16*)(ws + 32768 + 1048576);            // 4.5 MB
  _Float16* xm        = (_Float16*)(ws + 32768 + 1048576 + 4718592);  // 35.7 MB

  style_kernel       <<<16,   256, 0, stream>>>(sb, aw, ab, style_std);
  prep_w_kernel      <<<1024, 256, 0, stream>>>(cw, wsq, wk);
  demod_kernel       <<<16,   256, 0, stream>>>(wsq, style_std, demod);
  upsample_mod_kernel<<<BATCH * PADW * PADW, 128, 0, stream>>>(x, style_std, xm);
  conv_wmma_kernel   <<<512,  256, 0, stream>>>(wk, xm, demod, cb, ns, noise, t, pa, yout);
  torgb_kernel       <<<128,  256, 0, stream>>>(yout, ow, ob, out);
}
